// PhasedLSTM_28381143891990
// MI455X (gfx1250) — compile-verified, matrix operand order checked by reading, not run on hardware
//
#include <hip/hip_runtime.h>

// ---------------------------------------------------------------------------
// PhasedLSTM fused persistent-scan kernel for gfx1250 (MI455X, wave32, WMMA)
// Round 3: fix TDM builtin arity (6-arg toolchain variant)
// ---------------------------------------------------------------------------

typedef __bf16 bf16;
typedef __attribute__((ext_vector_type(16))) __bf16 v16bf;
typedef __attribute__((ext_vector_type(8)))  float  v8f;
typedef unsigned int u32x4 __attribute__((ext_vector_type(4)));
typedef int          i32x8 __attribute__((ext_vector_type(8)));
typedef int          i32x4 __attribute__((ext_vector_type(4)));

#define B_    64
#define T_    1024
#define D_    512
#define U_    512
#define FOURU 2048
#define ALPHA_ 0.001f

// packed sizes (elements)
#define XP_ELEMS  ((size_t)T_ * 4 * 16 * 32 * 16)   // 33,554,432
#define WP_ELEMS  ((size_t)128 * 16 * 32 * 16)      // 1,048,576
#define H_ELEMS   (4 * 16 * 32 * 16)                // 32,768 (one h buffer)
#define XP_BYTES  (XP_ELEMS * 2)
#define WP_BYTES  (WP_ELEMS * 2)
#define WSLICE_ELEMS (2 * 16 * 32 * 16)             // 16,384 bf16 = 32KB per WG

#if defined(__has_builtin)
#if __has_builtin(__builtin_amdgcn_tensor_load_to_lds) && __has_builtin(__builtin_amdgcn_s_wait_tensorcnt)
#define HAVE_TDM 1
#endif
#endif

// ---------------------------------------------------------------------------
// Prepass: pack x [B,T,D] f32 -> bf16 A-fragment layout (per timestep tiles)
// A-frag mapping: lane = m + 16*half ; kk = kk0 + 8*half ; kk0 = (j<8)?j:j+8
//                 linear = ((mtile*16 + ktile)*32 + lane)*16 + j
// ---------------------------------------------------------------------------
__global__ void plstm_pack_x(const float* __restrict__ x, bf16* __restrict__ xp) {
    const size_t o  = (size_t)blockIdx.x * blockDim.x + threadIdx.x;
    const int j     = (int)(o & 15);
    const int lane  = (int)((o >> 4) & 31);
    const int ktile = (int)((o >> 9) & 15);
    const int mt    = (int)((o >> 13) & 3);
    const int t     = (int)(o >> 15);
    const int m     = lane & 15;
    const int half  = lane >> 4;
    const int kk0   = (j < 8) ? j : j + 8;
    const int d     = ktile * 32 + kk0 + 8 * half;
    const int b     = mt * 16 + m;
    xp[o] = (bf16)x[((size_t)b * T_ + t) * D_ + d];
}

// ---------------------------------------------------------------------------
// Prepass: pack Wi / Wrec [512,2048] f32 -> bf16 B-fragment layout with
// gate-interleaved column permutation: col' = u*4 + g  (orig col = g*512 + u)
// ---------------------------------------------------------------------------
__global__ void plstm_pack_w(const float* __restrict__ wi, const float* __restrict__ wr,
                             bf16* __restrict__ wip, bf16* __restrict__ wrp) {
    const size_t o  = (size_t)blockIdx.x * blockDim.x + threadIdx.x;
    const float* src = blockIdx.y ? wr : wi;
    bf16*        dst = blockIdx.y ? wrp : wip;
    const int j     = (int)(o & 15);
    const int lane  = (int)((o >> 4) & 31);
    const int ktile = (int)((o >> 9) & 15);
    const int ntile = (int)(o >> 13);
    const int nn    = ntile * 16 + (lane & 15);
    const int half  = lane >> 4;
    const int kk0   = (j < 8) ? j : j + 8;
    const int k     = ktile * 32 + kk0 + 8 * half;
    const int u     = nn >> 2;
    const int g     = nn & 3;
    dst[o] = (bf16)src[(size_t)k * FOURU + g * U_ + u];
}

// ---------------------------------------------------------------------------
// Prepass: permuted bias, zero h double-buffer, reset grid-barrier counter
// ---------------------------------------------------------------------------
__global__ void plstm_init(const float* __restrict__ bias, float* __restrict__ biasp,
                           bf16* __restrict__ hglob, unsigned* __restrict__ ctr) {
    const int gid = blockIdx.x * blockDim.x + threadIdx.x;
    if (gid < FOURU)       biasp[gid] = bias[(gid & 3) * U_ + (gid >> 2)];
    if (gid < 2 * H_ELEMS) hglob[gid] = (bf16)0.f;
    if (gid == 0)          *ctr = 0u;
}

// ---------------------------------------------------------------------------
// Input projection (x_t @ Wi + bias) for one wave tile, dual WMMA chains.
// Independent of h -> issued in the barrier-wait shadow.
// ---------------------------------------------------------------------------
__device__ __forceinline__ void input_proj(const bf16* __restrict__ xp,
                                           const bf16* __restrict__ wip,
                                           int t, int mt, int ntg, int lane,
                                           float bcol, v8f& aca, v8f& acb)
{
    const bf16* __restrict__ xpt = xp + (size_t)t * 32768;
    #pragma unroll
    for (int r = 0; r < 8; ++r) { aca[r] = bcol; acb[r] = 0.f; }
    #pragma unroll
    for (int kt = 0; kt < 16; kt += 2) {
        const v16bf a0 = *(const v16bf*)(xpt + (((mt * 16 + kt    ) * 32 + lane) << 4));
        const v16bf b0 = *(const v16bf*)(wip + ((((size_t)ntg * 16 + kt    ) * 32 + lane) << 4));
        const v16bf a1 = *(const v16bf*)(xpt + (((mt * 16 + kt + 1) * 32 + lane) << 4));
        const v16bf b1 = *(const v16bf*)(wip + ((((size_t)ntg * 16 + kt + 1) * 32 + lane) << 4));
        aca = __builtin_amdgcn_wmma_f32_16x16x32_bf16(false, a0, false, b0, (short)0, aca, false, false);
        acb = __builtin_amdgcn_wmma_f32_16x16x32_bf16(false, a1, false, b1, (short)0, acb, false, false);
    }
}

// ---------------------------------------------------------------------------
// Persistent scan: 64 WGs x 256 threads (8 waves). Each WG owns 8 hidden
// units (32 permuted gate columns = 2 N-tiles, Wrec slice TDM-staged in LDS).
// Per wave per step: 16 recurrent WMMAs (critical path, LDS weights) +
// 16 input WMMAs (hidden behind the grid barrier). Grid barrier per step.
// ---------------------------------------------------------------------------
__global__ __launch_bounds__(256) void plstm_scan(
    const bf16* __restrict__ xp, const bf16* __restrict__ wip,
    const bf16* __restrict__ wrp, const float* __restrict__ biasp,
    const float* __restrict__ elapsed,
    const float* __restrict__ tau_p, const float* __restrict__ ron_p,
    const float* __restrict__ s_p,
    bf16* __restrict__ hglob, unsigned* __restrict__ ctr,
    float* __restrict__ out)
{
    __shared__ bf16  w_lds[WSLICE_ELEMS];  // 32 KB: this WG's Wrec slice
    __shared__ float z_lds[64 * 32];       // 8 KB: z slab [batch][local col]
    __shared__ float c_lds[512];           // persistent cell state  [b][j]
    __shared__ float h_lds[512];           // persistent hidden f32  [b][j]
    __shared__ float el_lds[64];           // elapsed[b] for current t

    const int tid  = threadIdx.x;
    const int lane = tid & 31;
    const int wv   = tid >> 5;
    const int mt   = wv & 3;               // M-tile (batch/16)
    const int ntl  = wv >> 2;              // local N-tile 0..1
    const int ntg  = blockIdx.x * 2 + ntl;

    c_lds[tid] = 0.f; c_lds[tid + 256] = 0.f;
    h_lds[tid] = 0.f; h_lds[tid + 256] = 0.f;

    // ---- stage this WG's Wrec slice (32 KB contiguous) into LDS ----
    const bf16* wr_src = wrp + (size_t)blockIdx.x * WSLICE_ELEMS;
#ifdef HAVE_TDM
    if (wv == 0) {
        const unsigned lds_off = (unsigned)(size_t)(&w_lds[0]);
        const unsigned long long gaddr = (unsigned long long)(size_t)wr_src;
        u32x4 g0;
        g0[0] = 1u;                                  // count=1, user descriptor
        g0[1] = lds_off;                             // lds_addr
        g0[2] = (unsigned)(gaddr & 0xffffffffull);   // global_addr[31:0]
        g0[3] = (unsigned)((gaddr >> 32) & 0x01ffffffull) | 0x80000000u; // type=2
        i32x8 g1;
        g1[0] = 0x00030000;                          // data_size=3 (8B), no mask
        g1[1] = (int)(4096u << 16);                  // tensor_dim0 = 4096 (lo16)
        g1[2] = (int)(1u << 16);                     // tensor_dim1 = 1
        g1[3] = (int)(4096u << 16);                  // tile_dim0 = 4096
        g1[4] = 0;                                   // tile_dim1/2 unused
        g1[5] = 4096;                                // tensor_dim0_stride
        g1[6] = 0; g1[7] = 0;
        i32x4 g2; g2[0] = 0; g2[1] = 0; g2[2] = 0; g2[3] = 0;
        i32x4 g3; g3[0] = 0; g3[1] = 0; g3[2] = 0; g3[3] = 0;
        i32x8 g4; g4[0] = 0; g4[1] = 0; g4[2] = 0; g4[3] = 0;
                  g4[4] = 0; g4[5] = 0; g4[6] = 0; g4[7] = 0;
        __builtin_amdgcn_tensor_load_to_lds(g0, g1, g2, g3, g4, 0);
        __builtin_amdgcn_s_wait_tensorcnt(0);
    }
#else
    for (int i = tid; i < WSLICE_ELEMS / 16; i += 256)
        *((v16bf*)&w_lds[i * 16]) = *((const v16bf*)&wr_src[i * 16]);
#endif

    const float tau_ = logf(1.f + expf(tau_p[0]));
    const float ron_ = logf(1.f + expf(ron_p[0]));
    const float s_   = logf(1.f + expf(s_p[0]));
    const float bcol = biasp[ntg * 16 + (lane & 15)];

    const int n    = lane & 15;
    const int half = lane >> 4;

    // input projection for t=0 (h_0 = 0; recurrent still runs on zero h)
    v8f aca, acb;
    input_proj(xp, wip, 0, mt, ntg, lane, bcol, aca, acb);

    __syncthreads();   // LDS staging + state init visible

    for (int t = 0; t < T_; ++t) {
        const bf16* __restrict__ hcur  = hglob + (size_t)(t & 1) * H_ELEMS;
        bf16* __restrict__       hnext = hglob + (size_t)((t + 1) & 1) * H_ELEMS;

        // ---- recurrent GEMM: h_t @ Wrec, weights from LDS (critical path) ----
        #pragma unroll
        for (int kt = 0; kt < 16; kt += 2) {
            const v16bf a0 = *(const v16bf*)(hcur  + (((mt * 16 + kt    ) * 32 + lane) << 4));
            const v16bf b0 = *(const v16bf*)(w_lds + (((ntl * 16 + kt    ) * 32 + lane) << 4));
            const v16bf a1 = *(const v16bf*)(hcur  + (((mt * 16 + kt + 1) * 32 + lane) << 4));
            const v16bf b1 = *(const v16bf*)(w_lds + (((ntl * 16 + kt + 1) * 32 + lane) << 4));
            aca = __builtin_amdgcn_wmma_f32_16x16x32_bf16(false, a0, false, b0, (short)0, aca, false, false);
            acb = __builtin_amdgcn_wmma_f32_16x16x32_bf16(false, a1, false, b1, (short)0, acb, false, false);
        }

        // merge chains, spill to LDS z-slab (C/D layout: lane n, row r+8*half)
        #pragma unroll
        for (int r = 0; r < 8; ++r) {
            const int m = mt * 16 + r + 8 * half;
            z_lds[m * 32 + ntl * 16 + n] = aca[r] + acb[r];
        }
        if (tid < 64) el_lds[tid] = elapsed[tid * T_ + t];
        __syncthreads();

        // ---- elementwise gates + time-gate k_t; 512 (batch,unit) items ----
        for (int idx = tid; idx < 512; idx += 256) {
            const int b = idx >> 3;
            const int j = idx & 7;
            const float z0 = z_lds[b * 32 + 4 * j + 0];
            const float z1 = z_lds[b * 32 + 4 * j + 1];
            const float z2 = z_lds[b * 32 + 4 * j + 2];
            const float z3 = z_lds[b * 32 + 4 * j + 3];

            float rmod = fmodf(el_lds[b] - s_, tau_);
            if (rmod < 0.f) rmod += tau_;           // python-style mod
            const float phit = rmod / tau_;
            float ktv;
            if (phit < 0.5f * ron_)  ktv = 2.f * phit * ron_;
            else if (phit < ron_)    ktv = 2.f - 2.f * phit / ron_;
            else                     ktv = ALPHA_ * phit;

            const float cprev = c_lds[idx];
            const float hprev = h_lds[idx];
            const float ia    = tanhf(z0);
            const float ig    = 1.f / (1.f + expf(-z1));
            const float fg    = 1.f / (1.f + expf(-(z2 + 1.f)));
            const float og    = 1.f / (1.f + expf(-z3));
            const float ct    = cprev * fg + ia * ig;
            const float cn    = ktv * ct + (1.f - ktv) * cprev;
            const float htld  = tanhf(ct) * og;
            const float hn    = ktv * htld + (1.f - ktv) * hprev;
            c_lds[idx] = cn;
            h_lds[idx] = hn;

            const int u = blockIdx.x * 8 + j;
            out[((size_t)b * T_ + t) * U_ + u] = hn;

            // scatter bf16 h into next step's A-fragment layout
            const int mt2 = b >> 4, mm = b & 15;
            const int kt2 = u >> 5, kk = u & 31;
            const int h2  = (kk >> 3) & 1;
            const int kk0 = kk - 8 * h2;
            const int jj  = (kk0 < 8) ? kk0 : kk0 - 8;
            const int ln2 = mm + 16 * h2;
            hnext[((mt2 * 16 + kt2) * 32 + ln2) * 16 + jj] = (bf16)hn;
        }

        __threadfence();          // release h writes device-wide
        __syncthreads();
        if (tid == 0) atomicAdd(ctr, 1u);   // arrive

        // ---- barrier shadow: input projection for t+1 (independent of h) ----
        if (t + 1 < T_) {
            if (t + 2 < T_)
                __builtin_prefetch(xp + (size_t)(t + 2) * 32768 + (size_t)tid * 128, 0, 1);
            input_proj(xp, wip, t + 1, mt, ntg, lane, bcol, aca, acb);
        }

        if (tid == 0) {           // wait
            const unsigned target = (unsigned)(t + 1) * gridDim.x;
            volatile unsigned* vc = ctr;
            while (*vc < target) { __builtin_amdgcn_s_sleep(1); }
        }
        __syncthreads();
    }
}

// ---------------------------------------------------------------------------
extern "C" void kernel_launch(void* const* d_in, const int* in_sizes, int n_in,
                              void* d_out, int out_size, void* d_ws, size_t ws_size,
                              hipStream_t stream) {
    const float* x       = (const float*)d_in[0];
    const float* elapsed = (const float*)d_in[1];
    const float* wi      = (const float*)d_in[2];
    const float* wr      = (const float*)d_in[3];
    const float* bias    = (const float*)d_in[4];
    const float* tau     = (const float*)d_in[5];
    const float* ron     = (const float*)d_in[6];
    const float* s       = (const float*)d_in[7];
    float* out           = (float*)d_out;

    char* ws = (char*)d_ws;
    bf16*     xp    = (bf16*)(ws);
    bf16*     wip   = (bf16*)(ws + XP_BYTES);
    bf16*     wrp   = (bf16*)(ws + XP_BYTES + WP_BYTES);
    float*    biasp = (float*)(ws + XP_BYTES + 2 * WP_BYTES);
    bf16*     hglob = (bf16*)(ws + XP_BYTES + 2 * WP_BYTES + 8192);
    unsigned* ctr   = (unsigned*)(ws + XP_BYTES + 2 * WP_BYTES + 8192 + 2 * H_ELEMS * 2);

    plstm_init<<<256, 256, 0, stream>>>(bias, biasp, hglob, ctr);
    plstm_pack_x<<<(unsigned)(XP_ELEMS / 256), 256, 0, stream>>>(x, xp);
    plstm_pack_w<<<dim3((unsigned)(WP_ELEMS / 256), 2), 256, 0, stream>>>(wi, wr, wip, wrp);
    plstm_scan<<<64, 256, 0, stream>>>(xp, wip, wrp, biasp, elapsed,
                                       tau, ron, s, hglob, ctr, out);
}
